// CxAM_45827301048702
// MI455X (gfx1250) — matrix operand order repaired and checked by reading, not compiled, not run on hardware
//
#include <hip/hip_runtime.h>
#include <cstdint>
#include <cstddef>

typedef __attribute__((ext_vector_type(16))) _Float16 v16h;
typedef __attribute__((ext_vector_type(8)))  _Float16 v8h;
typedef __attribute__((ext_vector_type(8)))  float    v8f;

static constexpr int B_ = 8, C_ = 512, T_ = 2048, CQ_ = 64;

// ---- workspace layout (bytes) ----
static constexpr size_t OFF_XT  = 0;                               // xT  [B][T][C]  f16
static constexpr size_t OFF_VTH = OFF_XT  + (size_t)B_*T_*C_*2;    // vT  [B][C][T]  f16
static constexpr size_t OFF_QH  = OFF_VTH + (size_t)B_*C_*T_*2;    // q   [B][T][CQ] f16
static constexpr size_t OFF_KH  = OFF_QH  + (size_t)B_*T_*CQ_*2;   // kT  [B][T][CQ] f16
static constexpr size_t OFF_WQ  = OFF_KH  + (size_t)B_*T_*CQ_*2;   // Wq f16
static constexpr size_t OFF_WK  = OFF_WQ  + (size_t)CQ_*C_*2;      // Wk f16
static constexpr size_t OFF_WV  = OFF_WK  + (size_t)CQ_*C_*2;      // Wv f16

__device__ __forceinline__ v16h mk16(v8h a0, v8h a1) {
  return __builtin_shufflevector(a0, a1, 0,1,2,3,4,5,6,7,8,9,10,11,12,13,14,15);
}
__device__ __forceinline__ v8f wmma16(v16h a, v16h b, v8f c) {
  return __builtin_amdgcn_wmma_f32_16x16x32_f16(false, a, false, b, (short)0, c, false, false);
}
// A fragment (16x32 f16, M-major rows): lane holds row M=lane%16,
// K = half*16+{0..15}  -> two contiguous 16B chunks.
__device__ __forceinline__ v16h a_frag(const _Float16* row, int kbase, int half) {
  return mk16(*(const v8h*)(row + kbase + half*16),
              *(const v8h*)(row + kbase + half*16 + 8));
}
// B fragment (32x16 f16, N-major rows): lane holds col N=lane%16,
// K = half*16+{0..15} -> one contiguous 32B chunk.

// ---------- 1) transpose x [B,C,T] f32 -> xT [B,T,C] f16 ----------
__global__ void k_transpose(const float* __restrict__ x, _Float16* __restrict__ xT) {
  __shared__ _Float16 tile[32][33];
  int t0 = blockIdx.x * 32, c0 = blockIdx.y * 32, b = blockIdx.z;
  int tx = threadIdx.x, ty = threadIdx.y;
#pragma unroll
  for (int ii = 0; ii < 4; ++ii) {
    int c = c0 + ty + 8*ii;
    tile[ty + 8*ii][tx] = (_Float16)x[((size_t)b*C_ + c)*T_ + t0 + tx];
  }
  __syncthreads();
#pragma unroll
  for (int ii = 0; ii < 4; ++ii) {
    int t = t0 + ty + 8*ii;
    xT[((size_t)b*T_ + t)*C_ + c0 + tx] = tile[tx][ty + 8*ii];
  }
}

// ---------- 2) weights f32 -> f16 ----------
__global__ void k_wconv(const float* __restrict__ Wq, const float* __restrict__ Wk,
                        const float* __restrict__ Wv, _Float16* __restrict__ Wq16,
                        _Float16* __restrict__ Wk16, _Float16* __restrict__ Wv16) {
  int i = blockIdx.x * 256 + threadIdx.x;
  if (i < CQ_*C_)                   Wq16[i]            = (_Float16)Wq[i];
  else if (i < 2*CQ_*C_)            Wk16[i - CQ_*C_]   = (_Float16)Wk[i - CQ_*C_];
  else if (i < 2*CQ_*C_ + C_*C_)    Wv16[i - 2*CQ_*C_] = (_Float16)Wv[i - 2*CQ_*C_];
}

// ---------- 3) q/k projection: Y[b,t,o] = sum_c W[o,c] x[b,c,t] + bias[o] ----------
__global__ void __launch_bounds__(256) k_qkproj(
    const _Float16* __restrict__ xT,
    const _Float16* __restrict__ Wq16, const float* __restrict__ bq,
    const _Float16* __restrict__ Wk16, const float* __restrict__ bk,
    _Float16* __restrict__ qh, _Float16* __restrict__ kh) {
  int wave = threadIdx.x >> 5, lane = threadIdx.x & 31;
  int half = lane >> 4, lr = lane & 15;
  int gw = blockIdx.x * 8 + wave;     // 8192 waves: b(8) x tt(128) x ot(4) x which(2)
  int b = gw >> 10;
  int rem = gw & 1023;
  int tt = rem >> 3;
  int ot = (rem >> 1) & 3;
  int which = rem & 1;
  const _Float16* W16 = which ? Wk16 : Wq16;
  const float*   bias = which ? bk : bq;
  _Float16*      Y    = which ? kh : qh;
  int t0 = tt * 16, o0 = ot * 16;
  const _Float16* arow = xT + ((size_t)b*T_ + t0 + lr) * C_;   // A: xT rows (M=t)
  const _Float16* brow = W16 + (size_t)(o0 + lr) * C_;         // B: W rows  (N=o)
  v8f acc = {};
#pragma unroll
  for (int kc = 0; kc < C_; kc += 32) {
    v16h a   = a_frag(arow, kc, half);
    v16h bfr = *(const v16h*)(brow + kc + half*16);
    acc = wmma16(a, bfr, acc);
  }
  float bval = bias[o0 + lr];
#pragma unroll
  for (int i = 0; i < 8; ++i) {
    int t = t0 + i + 8*half;
    Y[((size_t)b*T_ + t)*CQ_ + o0 + lr] = (_Float16)(acc[i] + bval);
  }
}

// ---------- 4) v projection, stored transposed: vT[b,c,t] = sum_c' Wv[c,c'] x[b,c',t] + bv[c] ----------
__global__ void __launch_bounds__(256) k_vproj(
    const _Float16* __restrict__ xT, const _Float16* __restrict__ Wv16,
    const float* __restrict__ bv, _Float16* __restrict__ vth) {
  int wave = threadIdx.x >> 5, lane = threadIdx.x & 31;
  int half = lane >> 4, lr = lane & 15;
  int gw = blockIdx.x * 8 + wave;     // 32768 waves: b(8) x ct(32) x tt(128)
  int b = gw >> 12;
  int rem = gw & 4095;
  int ct = rem >> 7;
  int tt = rem & 127;
  int c0 = ct * 16, t0 = tt * 16;
  const _Float16* arow = Wv16 + (size_t)(c0 + lr) * C_;        // A: Wv rows (M=c)
  const _Float16* brow = xT + ((size_t)b*T_ + t0 + lr) * C_;   // B: xT rows (N=t)
  v8f acc = {};
#pragma unroll
  for (int kc = 0; kc < C_; kc += 32) {
    v16h a   = a_frag(arow, kc, half);
    v16h bfr = *(const v16h*)(brow + kc + half*16);
    acc = wmma16(a, bfr, acc);
  }
#pragma unroll
  for (int i = 0; i < 8; ++i) {
    int c = c0 + i + 8*half;
    vth[((size_t)b*C_ + c)*T_ + t0 + lr] = (_Float16)(acc[i] + bv[c]);
  }
}

// ---------- 5) flash attention: out[b,c,t] = softmax_s(q·kT)[t,s] @ v[s,c] ----------
__global__ void __launch_bounds__(256) k_attn(
    const _Float16* __restrict__ qh, const _Float16* __restrict__ kh,
    const _Float16* __restrict__ vth, float* __restrict__ out) {
  __shared__ float    Pf[16][64];       // raw scores for this s-block
  __shared__ _Float16 Ph[16][64];       // exp(scores) f16
  __shared__ float    mrow[16], lrow[16], arw[16];
  int wave = threadIdx.x >> 5, lane = threadIdx.x & 31;
  int half = lane >> 4, lr = lane & 15;
  int b  = blockIdx.x >> 7;
  int t0 = (blockIdx.x & 127) * 16;
  if (threadIdx.x < 16) { mrow[threadIdx.x] = -3.0e38f; lrow[threadIdx.x] = 0.0f; }
  // q fragments for this 16-row query tile (K = CQ = 64 -> 2 steps), kept in regs
  const _Float16* qrow = qh + ((size_t)b*T_ + t0 + lr) * CQ_;
  v16h qa0 = a_frag(qrow, 0, half);
  v16h qa1 = a_frag(qrow, 32, half);
  v8f acc[4] = {};                      // wave owns out cols [wave*64, wave*64+64)
  __syncthreads();

  for (int sb = 0; sb < T_; sb += 64) {
    // --- scores: waves 0-3 each compute one 16x16 R subtile ---
    if (wave < 4) {
      const _Float16* krow = kh + ((size_t)b*T_ + sb + wave*16 + lr) * CQ_;
      v8f r = {};
      r = wmma16(qa0, *(const v16h*)(krow + half*16), r);
      r = wmma16(qa1, *(const v16h*)(krow + 32 + half*16), r);
#pragma unroll
      for (int i = 0; i < 8; ++i) Pf[i + 8*half][wave*16 + lr] = r[i];
    } else {
      int sn = sb + 64;                 // idle waves prefetch next K block
      if (sn < T_) {
        const _Float16* pf = kh + ((size_t)b*T_ + sn + (wave-4)*16 + lr) * CQ_;
        __builtin_prefetch(pf, 0, 1);
      }
    }
    __syncthreads();
    // --- online softmax, one thread per query row ---
    if (threadIdx.x < 16) {
      int rr = threadIdx.x;
      float mx = mrow[rr];
#pragma unroll 8
      for (int j = 0; j < 64; ++j) mx = fmaxf(mx, Pf[rr][j]);
      float alpha = __expf(mrow[rr] - mx);
      float ssum = 0.0f;
#pragma unroll 8
      for (int j = 0; j < 64; ++j) {
        float p = __expf(Pf[rr][j] - mx);
        Ph[rr][j] = (_Float16)p;
        ssum += p;
      }
      lrow[rr] = lrow[rr] * alpha + ssum;
      mrow[rr] = mx;
      arw[rr]  = alpha;
    }
    __syncthreads();
    // --- rescale accumulators, then P @ V on all 8 waves ---
    float al[8];
#pragma unroll
    for (int i = 0; i < 8; ++i) al[i] = arw[i + 8*half];
#pragma unroll
    for (int tile = 0; tile < 4; ++tile)
#pragma unroll
      for (int i = 0; i < 8; ++i) acc[tile][i] *= al[i];
    // P as A fragments from LDS (row M = lane%16)
    v16h pa0, pa1;
#pragma unroll
    for (int e = 0; e < 8; ++e) {
      pa0[e]   = Ph[lr][half*16 + e];
      pa0[8+e] = Ph[lr][8 + half*16 + e];
      pa1[e]   = Ph[lr][32 + half*16 + e];
      pa1[8+e] = Ph[lr][40 + half*16 + e];
    }
    int c0 = wave * 64;
#pragma unroll
    for (int tile = 0; tile < 4; ++tile) {
      const _Float16* vrow = vth + ((size_t)b*C_ + c0 + tile*16 + lr) * T_ + sb;
      acc[tile] = wmma16(pa0, *(const v16h*)(vrow + half*16), acc[tile]);
      acc[tile] = wmma16(pa1, *(const v16h*)(vrow + 32 + half*16), acc[tile]);
    }
    __syncthreads();                    // protect Pf/Ph for next s-block
  }
  // --- normalize and store transposed output [B,C,T] f32 ---
  float li[8];
#pragma unroll
  for (int i = 0; i < 8; ++i) li[i] = 1.0f / lrow[i + 8*half];
#pragma unroll
  for (int tile = 0; tile < 4; ++tile) {
    int c = wave*64 + tile*16 + lr;
#pragma unroll
    for (int i = 0; i < 8; ++i) {
      int t = t0 + i + 8*half;
      out[((size_t)b*C_ + c)*T_ + t] = acc[tile][i] * li[i];
    }
  }
}

extern "C" void kernel_launch(void* const* d_in, const int* in_sizes, int n_in,
                              void* d_out, int out_size, void* d_ws, size_t ws_size,
                              hipStream_t stream) {
  const float* x  = (const float*)d_in[0];
  const float* Wq = (const float*)d_in[1];
  const float* bq = (const float*)d_in[2];
  const float* Wk = (const float*)d_in[3];
  const float* bk = (const float*)d_in[4];
  const float* Wv = (const float*)d_in[5];
  const float* bv = (const float*)d_in[6];
  float* out = (float*)d_out;
  char* ws = (char*)d_ws;
  _Float16* xT   = (_Float16*)(ws + OFF_XT);
  _Float16* vth  = (_Float16*)(ws + OFF_VTH);
  _Float16* qh   = (_Float16*)(ws + OFF_QH);
  _Float16* kh   = (_Float16*)(ws + OFF_KH);
  _Float16* Wq16 = (_Float16*)(ws + OFF_WQ);
  _Float16* Wk16 = (_Float16*)(ws + OFF_WK);
  _Float16* Wv16 = (_Float16*)(ws + OFF_WV);

  dim3 tb(32, 8, 1), tg(T_/32, C_/32, B_);
  k_transpose<<<tg, tb, 0, stream>>>(x, xT);
  int wtot = 2*CQ_*C_ + C_*C_;
  k_wconv<<<(wtot + 255)/256, 256, 0, stream>>>(Wq, Wk, Wv, Wq16, Wk16, Wv16);
  k_qkproj<<<1024, 256, 0, stream>>>(xT, Wq16, bq, Wk16, bk, qh, kh);
  k_vproj<<<4096, 256, 0, stream>>>(xT, Wv16, bv, vth);
  k_attn<<<1024, 256, 0, stream>>>(qh, kh, vth, out);
}
